// ProbabilisticGraphNetwork_72370198938041
// MI455X (gfx1250) — matrix operand order repaired and checked by reading, not compiled, not run on hardware
//
#include <hip/hip_runtime.h>
#include <hip/hip_bf16.h>

typedef __attribute__((ext_vector_type(16))) _Float16 v16h;
typedef __attribute__((ext_vector_type(8)))  _Float16 v8h;
typedef __attribute__((ext_vector_type(8)))  float    v8f;

#define OUTC 256
#define KDIM 512          // 2*OUTC
#define EDGE_WAVES 4      // waves per block in edge kernel
#define NODES_PER_BLOCK 8

// ---------------------------------------------------------------------------
// h = act @ W_lin^T + b ; written as f32 (for messages) and f16 (WMMA A side)
// ---------------------------------------------------------------------------
__global__ void linear_kernel(const float* __restrict__ act,
                              const float* __restrict__ W,
                              const float* __restrict__ b,
                              float* __restrict__ h32,
                              _Float16* __restrict__ h16,
                              int nNodes, int inC) {
  __shared__ float xs[NODES_PER_BLOCK][256];
  const int t = threadIdx.x;                    // 256 threads = out channel
  const int nb = blockIdx.x * NODES_PER_BLOCK;
  for (int r = 0; r < NODES_PER_BLOCK; ++r) {
    int n = nb + r;
    if (t < inC) xs[r][t] = (n < nNodes) ? act[(size_t)n * inC + t] : 0.f;
  }
  __syncthreads();
  float sum[NODES_PER_BLOCK];
  const float bias = b[t];
#pragma unroll
  for (int r = 0; r < NODES_PER_BLOCK; ++r) sum[r] = bias;
  for (int k = 0; k < inC; ++k) {
    float w = W[(size_t)t * inC + k];
#pragma unroll
    for (int r = 0; r < NODES_PER_BLOCK; ++r) sum[r] = fmaf(w, xs[r][k], sum[r]);
  }
  for (int r = 0; r < NODES_PER_BLOCK; ++r) {
    int n = nb + r;
    if (n < nNodes) {
      h32[(size_t)n * OUTC + t] = sum[r];
      h16[(size_t)n * OUTC + t] = (_Float16)sum[r];
    }
  }
}

// ---------------------------------------------------------------------------
// Pre-swizzle W_s1 (OUTC x KDIM, row major, f32) into B-fragment layout f16:
// Wsw[((kc*16 + nt)*32 + lane)*16 + j] = W_s1[nt*16 + (lane&15)][kc*32 + ((lane>=16)?16:0) + j]
// so each B fragment is 32 contiguous bytes per lane.
// ---------------------------------------------------------------------------
__global__ void swizzle_kernel(const float* __restrict__ Ws1,
                               _Float16* __restrict__ Wsw) {
  int tid = blockIdx.x * 256 + threadIdx.x;     // 16*16*32*16 = 131072 total
  if (tid >= 16 * 16 * 32 * 16) return;
  int j  = tid & 15;
  int l  = (tid >> 4) & 31;
  int nt = (tid >> 9) & 15;
  int kc = tid >> 13;
  int chan = nt * 16 + (l & 15);
  int K    = kc * 32 + ((l >> 4) ? 16 : 0) + j;
  Wsw[tid] = (_Float16)Ws1[(size_t)chan * KDIM + K];
}

__global__ void relu_kernel(const float* __restrict__ in,
                            float* __restrict__ out, int n) {
  int i = blockIdx.x * 256 + threadIdx.x;
  if (i < n) out[i] = fmaxf(in[i], 0.f);
}

// ---------------------------------------------------------------------------
// Fused edge MLP + score + scatter-add. One wave per 16-edge tile.
//   cat  = [h[dst] | h[src]]            (16 x 512, f16, staged in LDS)
//   hid  = relu(cat @ W_s1^T + b_s1)    (16 x 256, f32 acc via WMMA f16)
//   sc   = sigmoid(hid . w_s2 + b_s2)   (16)
//   accum[dst] += sc * h32[src]         (atomic f32)
// n-tiles processed in groups of 4 so only 4 v8f accumulators (32 VGPRs)
// are ever live -> no scratch spills of the WMMA accumulators.
// ---------------------------------------------------------------------------
__global__ void __launch_bounds__(EDGE_WAVES * 32)
edge_kernel(const int* __restrict__ srcIdx,
            const int* __restrict__ dstIdx,
            const _Float16* __restrict__ h16,
            const float* __restrict__ h32,
            const _Float16* __restrict__ Wsw,
            const float* __restrict__ bs1,
            const float* __restrict__ ws2,
            const float* __restrict__ bs2,
            float* __restrict__ accum,
            int E, int nNodes) {
  __shared__ _Float16 catL[EDGE_WAVES][16][KDIM];   // 64 KB
  __shared__ int   sArr[EDGE_WAVES][16];
  __shared__ int   dArr[EDGE_WAVES][16];
  __shared__ float scoreL[EDGE_WAVES][16];

  const int lane  = threadIdx.x & 31;
  const int w     = threadIdx.x >> 5;
  const int tile  = blockIdx.x * EDGE_WAVES + w;
  const int base  = tile * 16;
  const int total = E + nNodes;                 // self loops appended

  if (lane < 16) {
    int ge = base + lane;
    int s = 0, d = 0;
    if (ge < E)            { s = srcIdx[ge]; d = dstIdx[ge]; }
    else if (ge < total)   { s = ge - E;     d = s; }
    sArr[w][lane] = s;
    dArr[w][lane] = d;
  }
  __syncthreads();

  // Gather: per edge, 256 f16 of xi then 256 f16 of xj; 32 lanes x 16B each.
  for (int e = 0; e < 16; ++e) {
    int d = dArr[w][e], s = sArr[w][e];
    const uint4* pi = (const uint4*)(h16 + (size_t)d * OUTC) + lane;
    const uint4* pj = (const uint4*)(h16 + (size_t)s * OUTC) + lane;
    *((uint4*)(&catL[w][e][0])    + lane) = *pi;
    *((uint4*)(&catL[w][e][OUTC]) + lane) = *pj;
  }
  __syncthreads();

  const int m  = lane & 15;                 // A row / C column
  const int kb = (lane >> 4) ? 16 : 0;      // A K-half select

  float part[8];
#pragma unroll
  for (int v = 0; v < 8; ++v) part[v] = 0.f;

  v8f zero = {};

#pragma clang loop unroll(disable)
  for (int g = 0; g < 4; ++g) {             // n-tile groups of 4
    v8f acc[4];
#pragma unroll
    for (int q = 0; q < 4; ++q) acc[q] = zero;

    for (int kc = 0; kc < 16; ++kc) {
      const _Float16* arow = &catL[w][m][kc * 32 + kb];
      v8h lo = *(const v8h*)(arow);
      v8h hi = *(const v8h*)(arow + 8);
      v16h a;
#pragma unroll
      for (int j = 0; j < 8; ++j) { a[j] = lo[j]; a[8 + j] = hi[j]; }
      const _Float16* bbase =
          Wsw + (((size_t)kc * 16 + g * 4) * 32 + lane) * 16;
#pragma unroll
      for (int q = 0; q < 4; ++q) {
        v16h bfrag = *(const v16h*)(bbase + (size_t)q * 32 * 16);
        acc[q] = __builtin_amdgcn_wmma_f32_16x16x32_f16(
            false, a, false, bfrag, (short)0, acc[q], false, false);
      }
    }

    // bias + ReLU + fold this group's scoring-dot contribution
#pragma unroll
    for (int q = 0; q < 4; ++q) {
      int chan   = (g * 4 + q) * 16 + m;
      float bias = bs1[chan];
      float wv   = ws2[chan];
#pragma unroll
      for (int v = 0; v < 8; ++v) {
        float hv = fmaxf(acc[q][v] + bias, 0.f);
        part[v] = fmaf(hv, wv, part[v]);
      }
    }
  }

#pragma unroll
  for (int v = 0; v < 8; ++v) {
    float p = part[v];
    p += __shfl_xor(p, 1, 16);
    p += __shfl_xor(p, 2, 16);
    p += __shfl_xor(p, 4, 16);
    p += __shfl_xor(p, 8, 16);
    part[v] = p;
  }
  float b2 = bs2[0];
  if (m == 0) {                              // lanes 0 (edges 0-7) and 16 (8-15)
    int rowoff = (lane >> 4) ? 8 : 0;
#pragma unroll
    for (int v = 0; v < 8; ++v)
      scoreL[w][rowoff + v] = 1.f / (1.f + __expf(-(part[v] + b2)));
  }
  __syncthreads();

  // scatter: accum[dst] += score * h32[src]
  for (int e = 0; e < 16; ++e) {
    int ge = base + e;
    if (ge >= total) break;
    float sc = scoreL[w][e];
    const float* xj = h32 + (size_t)sArr[w][e] * OUTC;
    float* outp = accum + (size_t)dArr[w][e] * OUTC;
#pragma unroll
    for (int cc = 0; cc < 8; ++cc) {
      int c = lane + cc * 32;
      atomicAdd(outp + c, sc * xj[c]);
    }
  }
}

// ---------------------------------------------------------------------------
extern "C" void kernel_launch(void* const* d_in, const int* in_sizes, int n_in,
                              void* d_out, int out_size, void* d_ws, size_t ws_size,
                              hipStream_t stream) {
  const float* x  = (const float*)d_in[0];
  const int*   ei = (const int*)d_in[1];
  const int N = in_sizes[0] / 64;
  const int E = in_sizes[1] / 2;
  const int* src = ei;
  const int* dst = ei + E;

  // workspace carve-out (~36 MB total)
  char* ws = (char*)d_ws;
  size_t off = 0;
  auto take = [&](size_t bytes) {
    char* p = ws + off;
    off = (off + bytes + 255) & ~(size_t)255;
    return p;
  };
  float*    h32    = (float*)   take((size_t)N * OUTC * 4);
  _Float16* h16    = (_Float16*)take((size_t)N * OUTC * 2);
  float*    accumW = (float*)   take((size_t)N * OUTC * 4);
  float*    act    = (float*)   take((size_t)N * OUTC * 4);
  _Float16* Wsw    = (_Float16*)take((size_t)16 * 16 * 32 * 16 * 2);
  (void)ws_size; (void)n_in; (void)out_size;

  const float* actIn = x;
  int inC = 64;
  const int nTiles = (E + N + 15) / 16;
  const int edgeBlocks = (nTiles + EDGE_WAVES - 1) / EDGE_WAVES;

  for (int l = 0; l < 3; ++l) {
    const float* Wlin = (const float*)d_in[2 + 6 * l + 0];
    const float* blin = (const float*)d_in[2 + 6 * l + 1];
    const float* Ws1  = (const float*)d_in[2 + 6 * l + 2];
    const float* bs1  = (const float*)d_in[2 + 6 * l + 3];
    const float* Ws2  = (const float*)d_in[2 + 6 * l + 4];
    const float* bs2  = (const float*)d_in[2 + 6 * l + 5];
    float* accum = (l == 2) ? (float*)d_out : accumW;

    linear_kernel<<<(N + NODES_PER_BLOCK - 1) / NODES_PER_BLOCK, 256, 0, stream>>>(
        actIn, Wlin, blin, h32, h16, N, inC);
    swizzle_kernel<<<(16 * 16 * 32 * 16) / 256, 256, 0, stream>>>(Ws1, Wsw);
    hipMemsetAsync(accum, 0, (size_t)N * OUTC * 4, stream);
    edge_kernel<<<edgeBlocks, EDGE_WAVES * 32, 0, stream>>>(
        src, dst, h16, h32, Wsw, bs1, Ws2, bs2, accum, E, N);
    if (l < 2) {
      relu_kernel<<<((N * OUTC) + 255) / 256, 256, 0, stream>>>(accum, act, N * OUTC);
      actIn = act;
      inC = OUTC;
    }
  }
}
